// NA3DBlock_69879117906012
// MI455X (gfx1250) — matrix operand order repaired and checked by reading, not compiled, not run on hardware
//
#include <hip/hip_runtime.h>
#include <math.h>

typedef __attribute__((ext_vector_type(16))) _Float16 v16h;
typedef __attribute__((ext_vector_type(8)))  _Float16 v8h;
typedef __attribute__((ext_vector_type(8)))  float    v8f;

#define NTOK   8000      // 20*20*20
#define CCH    128
#define NHEADS 8
#define HD     16
#define KWIN   5
#define SDIM   20

// ---------------------------------------------------------------------------
// Pack weight w [N,K] fp32 into WMMA B-fragment-major f16 layout:
//   packed[nt][ks][lane][h] = B[k][n] = w[n][k]
// with n = nt*16 + (lane&15), k = ks*32 + (lane>>4)*16 + h  (h = 0..15).
// Each lane's 16 halves (32B) are contiguous -> wave B-frag = 1 coalesced
// 32B load per k-step, no LDS, no scalar packing.
// ---------------------------------------------------------------------------
__global__ void pack_b_wmma(const float* __restrict__ w, _Float16* __restrict__ bp,
                            int N, int Kd) {
    int tid = blockIdx.x * blockDim.x + threadIdx.x;
    if (tid >= N * Kd) return;
    int h    = tid & 15;
    int lane = (tid >> 4) & 31;
    int nks  = Kd >> 5;
    int ks   = (tid >> 9) % nks;
    int nt   = tid / (nks << 9);
    int n = nt * 16 + (lane & 15);
    int k = ks * 32 + ((lane >> 4) * 16) + h;
    bp[tid] = (_Float16)w[(size_t)n * Kd + k];
}

// ---------------------------------------------------------------------------
// BatchNorm stats (training mode, biased var): one block per channel.
// x is channel-major [C, n]. Produces per-channel scale/shift.
// ---------------------------------------------------------------------------
__global__ __launch_bounds__(256) void bn_stats(const float* __restrict__ x,
                                                const float* __restrict__ gamma,
                                                const float* __restrict__ beta,
                                                float* __restrict__ scale,
                                                float* __restrict__ shift, int n) {
    int c = blockIdx.x;
    const float* p = x + (size_t)c * n;
    float s = 0.f, s2 = 0.f;
    for (int i = threadIdx.x; i < n; i += 256) { float v = p[i]; s += v; s2 += v * v; }
    __shared__ float sh0[256], sh1[256];
    sh0[threadIdx.x] = s; sh1[threadIdx.x] = s2;
    __syncthreads();
    for (int o = 128; o > 0; o >>= 1) {
        if (threadIdx.x < o) { sh0[threadIdx.x] += sh0[threadIdx.x + o];
                               sh1[threadIdx.x] += sh1[threadIdx.x + o]; }
        __syncthreads();
    }
    if (threadIdx.x == 0) {
        float mean = sh0[0] / (float)n;
        float var  = sh1[0] / (float)n - mean * mean;
        float sc   = gamma[c] * rsqrtf(var + 1e-5f);
        scale[c] = sc;
        shift[c] = beta[c] - mean * sc;
    }
}

// Apply BN and transpose channel-major [C,n] -> token-major f16 [n,C].
__global__ void bn_apply_f16(const float* __restrict__ x,
                             const float* __restrict__ scale,
                             const float* __restrict__ shift,
                             _Float16* __restrict__ h, int n) {
    int tid = blockIdx.x * blockDim.x + threadIdx.x;  // n*128 threads
    if (tid >= n * CCH) return;
    int c = tid & (CCH - 1), t = tid >> 7;
    h[tid] = (_Float16)(x[(size_t)c * n + t] * scale[c] + shift[c]);
}

// ---------------------------------------------------------------------------
// WMMA GEMM:  C[M,N] = A[M,K](f16, row-major) x B(+bias), B pre-swizzled to
// fragment-major layout (see pack_b_wmma). Block tile 64x64, K-step 32,
// 128 threads (4 waves). Wave w owns column tile w, iterates the 4 row tiles
// -> 4 x v_wmma_f32_16x16x32_f16 per k-step. A staged via LDS (ds_load_b128
// fragment reads), B read straight from global (L2-resident weights).
// EPI 0: fp32 out row-major            (qkv)
// EPI 1: f16 out row-major, exact GELU (mlp1)
// EPI 2: fp32 out TRANSPOSED [N,M] with fused residual add (proj / mlp2)
// Requires M%64==0, N%64==0, K%32==0 (true for all four GEMMs here).
// ---------------------------------------------------------------------------
template <int EPI>
__global__ __launch_bounds__(128) void gemm_wmma(
    const _Float16* __restrict__ A, const _Float16* __restrict__ Bp,
    const float* __restrict__ bias, int M, int N, int Kd,
    float* __restrict__ outF, _Float16* __restrict__ outH,
    const float* __restrict__ res) {
    __shared__ _Float16 lA[64 * 40];   // 64 rows x 32 K, pad to 40
    const int tid  = threadIdx.x;
    const int wave = tid >> 5;
    const int lane = tid & 31;
    const int bm = blockIdx.y * 64;
    const int bn = blockIdx.x * 64;
    const int nks = Kd >> 5;

    v8f acc[4] = {};

    const int arow = tid >> 1, acol = (tid & 1) * 16;   // A tile: 2 thr/row
    const int lrow = lane & 15;
    const int akb  = (lane >> 4) * 16;   // A frag K base (halves 0..15)

    // per-lane base of this wave's packed B fragments
    const _Float16* bbase =
        Bp + (((size_t)(blockIdx.x * 4 + wave) * nks) * 32 + lane) * 16;

    for (int ks = 0; ks < nks; ++ks) {
        // stage A tile (32B vector loads, 16B-aligned LDS stores)
        const uint4* gA = reinterpret_cast<const uint4*>(
            A + (size_t)(bm + arow) * Kd + ks * 32 + acol);
        uint4* sA = reinterpret_cast<uint4*>(&lA[arow * 40 + acol]);
        sA[0] = gA[0]; sA[1] = gA[1];

        // B fragment: one 32B coalesced global load per lane (32B-aligned)
        v16h bf = *reinterpret_cast<const v16h*>(bbase + (size_t)ks * 512);

        __syncthreads();
#pragma unroll
        for (int mi = 0; mi < 4; ++mi) {
            v8h lo = *reinterpret_cast<const v8h*>(&lA[(mi * 16 + lrow) * 40 + akb]);
            v8h hi = *reinterpret_cast<const v8h*>(&lA[(mi * 16 + lrow) * 40 + akb + 8]);
            v16h af;
#pragma unroll
            for (int h = 0; h < 8; ++h) { af[h] = lo[h]; af[h + 8] = hi[h]; }
            acc[mi] = __builtin_amdgcn_wmma_f32_16x16x32_f16(
                false, af, false, bf, (short)0, acc[mi], false, false);
        }
        __syncthreads();
    }

    // epilogue: lane holds D col = lane&15, rows (lane>>4)*8 + r
    const int col = bn + wave * 16 + lrow;
    const float bv = bias[col];
    const int rbase = (lane >> 4) * 8;
#pragma unroll
    for (int mi = 0; mi < 4; ++mi) {
#pragma unroll
        for (int r = 0; r < 8; ++r) {
            int mrow = bm + mi * 16 + rbase + r;
            float v = acc[mi][r] + bv;
            if (EPI == 0) {
                outF[(size_t)mrow * N + col] = v;
            } else if (EPI == 1) {
                float g = 0.5f * v * (1.0f + erff(v * 0.70710678118654752f));
                outH[(size_t)mrow * N + col] = (_Float16)g;
            } else {
                size_t o = (size_t)col * M + mrow;
                outF[o] = v + res[o];
            }
        }
    }
}

// ---------------------------------------------------------------------------
// 3D neighborhood attention: one thread per (token, head).
// qkv fp32 row-major [NTOK, 384] = [q(128) | k(128) | v(128)].
// Flash-style online softmax over the 5x5x5 clamped window + rpb bias.
// Output f16 token-major [NTOK, 128] (A-operand for the proj GEMM).
// ---------------------------------------------------------------------------
__global__ __launch_bounds__(256) void na3d_attn(const float* __restrict__ qkv,
                                                 const float* __restrict__ rpb,
                                                 _Float16* __restrict__ attn_out) {
    int id = blockIdx.x * blockDim.x + threadIdx.x;   // NTOK*NHEADS
    if (id >= NTOK * NHEADS) return;
    int t = id >> 3, n = id & 7;
    int z = t / (SDIM * SDIM);
    int rm = t % (SDIM * SDIM);
    int y = rm / SDIM, xw = rm % SDIM;

    auto start = [](int i) { int s = i - KWIN / 2;
                             if (s < 0) s = 0; if (s > SDIM - KWIN) s = SDIM - KWIN; return s; };
    int sz = start(z), sy = start(y), sx = start(xw);
    int rd0 = sz - z + (KWIN - 1);
    int rh0 = sy - y + (KWIN - 1);
    int rw0 = sx - xw + (KWIN - 1);

    const float* qp = qkv + (size_t)t * 384 + n * HD;
    float4 q0 = *(const float4*)(qp + 0);
    float4 q1 = *(const float4*)(qp + 4);
    float4 q2 = *(const float4*)(qp + 8);
    float4 q3 = *(const float4*)(qp + 12);
    const float qs = 0.25f;  // hd^-0.5, hd = 16
    q0.x *= qs; q0.y *= qs; q0.z *= qs; q0.w *= qs;
    q1.x *= qs; q1.y *= qs; q1.z *= qs; q1.w *= qs;
    q2.x *= qs; q2.y *= qs; q2.z *= qs; q2.w *= qs;
    q3.x *= qs; q3.y *= qs; q3.z *= qs; q3.w *= qs;

    float m = -3.0e38f, l = 0.f;
    float o[HD];
#pragma unroll
    for (int c = 0; c < HD; ++c) o[c] = 0.f;

    const float* rpbn = rpb + (size_t)n * 9 * 9 * 9;

    for (int iz = 0; iz < KWIN; ++iz) {
        for (int iy = 0; iy < KWIN; ++iy) {
            for (int ix = 0; ix < KWIN; ++ix) {
                int tt = (sz + iz) * (SDIM * SDIM) + (sy + iy) * SDIM + (sx + ix);
                const float* kp = qkv + (size_t)tt * 384 + CCH + n * HD;
                float4 k0 = *(const float4*)(kp + 0);
                float4 k1 = *(const float4*)(kp + 4);
                float4 k2 = *(const float4*)(kp + 8);
                float4 k3 = *(const float4*)(kp + 12);
                float s = q0.x * k0.x + q0.y * k0.y + q0.z * k0.z + q0.w * k0.w
                        + q1.x * k1.x + q1.y * k1.y + q1.z * k1.z + q1.w * k1.w
                        + q2.x * k2.x + q2.y * k2.y + q2.z * k2.z + q2.w * k2.w
                        + q3.x * k3.x + q3.y * k3.y + q3.z * k3.z + q3.w * k3.w;
                s += rpbn[((rd0 + iz) * 9 + (rh0 + iy)) * 9 + (rw0 + ix)];

                float mnew = fmaxf(m, s);
                float corr = __expf(m - mnew);
                float w = __expf(s - mnew);
                l = l * corr + w;
                m = mnew;
                const float* vp = qkv + (size_t)tt * 384 + 2 * CCH + n * HD;
                float4 v0 = *(const float4*)(vp + 0);
                float4 v1 = *(const float4*)(vp + 4);
                float4 v2 = *(const float4*)(vp + 8);
                float4 v3 = *(const float4*)(vp + 12);
                o[0]  = o[0]  * corr + w * v0.x;  o[1]  = o[1]  * corr + w * v0.y;
                o[2]  = o[2]  * corr + w * v0.z;  o[3]  = o[3]  * corr + w * v0.w;
                o[4]  = o[4]  * corr + w * v1.x;  o[5]  = o[5]  * corr + w * v1.y;
                o[6]  = o[6]  * corr + w * v1.z;  o[7]  = o[7]  * corr + w * v1.w;
                o[8]  = o[8]  * corr + w * v2.x;  o[9]  = o[9]  * corr + w * v2.y;
                o[10] = o[10] * corr + w * v2.z;  o[11] = o[11] * corr + w * v2.w;
                o[12] = o[12] * corr + w * v3.x;  o[13] = o[13] * corr + w * v3.y;
                o[14] = o[14] * corr + w * v3.z;  o[15] = o[15] * corr + w * v3.w;
            }
        }
    }
    float inv = 1.0f / l;
    _Float16* op = attn_out + (size_t)t * CCH + n * HD;
#pragma unroll
    for (int c = 0; c < HD; ++c) op[c] = (_Float16)(o[c] * inv);
}

// ---------------------------------------------------------------------------
extern "C" void kernel_launch(void* const* d_in, const int* in_sizes, int n_in,
                              void* d_out, int out_size, void* d_ws, size_t ws_size,
                              hipStream_t stream) {
    const float* x      = (const float*)d_in[0];   // [128, 8000] channel-major
    const float* gamma1 = (const float*)d_in[1];
    const float* beta1  = (const float*)d_in[2];
    const float* w_qkv  = (const float*)d_in[3];   // [384,128]
    const float* b_qkv  = (const float*)d_in[4];
    const float* rpb    = (const float*)d_in[5];   // [8,9,9,9]
    const float* w_proj = (const float*)d_in[6];   // [128,128]
    const float* b_proj = (const float*)d_in[7];
    const float* gamma2 = (const float*)d_in[8];
    const float* beta2  = (const float*)d_in[9];
    const float* w1     = (const float*)d_in[10];  // [512,128]
    const float* b1     = (const float*)d_in[11];
    const float* w2     = (const float*)d_in[12];  // [128,512]
    const float* b2     = (const float*)d_in[13];
    float* out = (float*)d_out;                    // [128, 8000]

    // workspace carve-out (256B aligned)
    char* ws = (char*)d_ws;
    size_t off = 0;
    auto carve = [&](size_t bytes) -> void* {
        void* p = ws + off;
        off = (off + bytes + 255) & ~(size_t)255;
        return p;
    };
    _Float16* bqkvP  = (_Float16*)carve(128 * 384 * 2);   // packed B for qkv
    _Float16* bprojP = (_Float16*)carve(128 * 128 * 2);   // packed B for proj
    _Float16* b1P    = (_Float16*)carve(128 * 512 * 2);   // packed B for mlp1
    _Float16* b2P    = (_Float16*)carve(512 * 128 * 2);   // packed B for mlp2
    float*    scale1 = (float*)carve(CCH * 4);
    float*    shift1 = (float*)carve(CCH * 4);
    float*    scale2 = (float*)carve(CCH * 4);
    float*    shift2 = (float*)carve(CCH * 4);
    _Float16* h1     = (_Float16*)carve((size_t)NTOK * CCH * 2);
    float*    qkv    = (float*)carve((size_t)NTOK * 384 * 4);
    _Float16* attn   = (_Float16*)carve((size_t)NTOK * CCH * 2);
    float*    x1     = (float*)carve((size_t)NTOK * CCH * 4);
    _Float16* h2     = (_Float16*)carve((size_t)NTOK * CCH * 2);
    _Float16* y1     = (_Float16*)carve((size_t)NTOK * 512 * 2);

    // 1) weights -> packed WMMA B-fragment layout (f16)
    pack_b_wmma<<<(384 * 128 + 255) / 256, 256, 0, stream>>>(w_qkv, bqkvP, 384, 128);
    pack_b_wmma<<<(128 * 128 + 255) / 256, 256, 0, stream>>>(w_proj, bprojP, 128, 128);
    pack_b_wmma<<<(512 * 128 + 255) / 256, 256, 0, stream>>>(w1, b1P, 512, 128);
    pack_b_wmma<<<(128 * 512 + 255) / 256, 256, 0, stream>>>(w2, b2P, 128, 512);

    // 2) BN1 -> token-major f16
    bn_stats<<<CCH, 256, 0, stream>>>(x, gamma1, beta1, scale1, shift1, NTOK);
    bn_apply_f16<<<(NTOK * CCH + 255) / 256, 256, 0, stream>>>(x, scale1, shift1, h1, NTOK);

    // 3) qkv GEMM: [8000,128] x [128,384] + b_qkv -> fp32 [8000,384]
    gemm_wmma<0><<<dim3(384 / 64, NTOK / 64), 128, 0, stream>>>(
        h1, bqkvP, b_qkv, NTOK, 384, 128, qkv, nullptr, nullptr);

    // 4) neighborhood attention -> f16 [8000,128]
    na3d_attn<<<(NTOK * NHEADS + 255) / 256, 256, 0, stream>>>(qkv, rpb, attn);

    // 5) proj GEMM with fused residual, transposed out: x1[c,t] = x + attn@W^T + b
    gemm_wmma<2><<<dim3(128 / 64, NTOK / 64), 128, 0, stream>>>(
        attn, bprojP, b_proj, NTOK, 128, 128, x1, nullptr, x);

    // 6) BN2 -> token-major f16
    bn_stats<<<CCH, 256, 0, stream>>>(x1, gamma2, beta2, scale2, shift2, NTOK);
    bn_apply_f16<<<(NTOK * CCH + 255) / 256, 256, 0, stream>>>(x1, scale2, shift2, h2, NTOK);

    // 7) MLP1: [8000,128] x [128,512] + b1, exact GELU -> f16 [8000,512]
    gemm_wmma<1><<<dim3(512 / 64, NTOK / 64), 128, 0, stream>>>(
        h2, b1P, b1, NTOK, 512, 128, nullptr, y1, nullptr);

    // 8) MLP2 with fused residual, transposed out -> final output [128,8000]
    gemm_wmma<2><<<dim3(128 / 64, NTOK / 64), 128, 0, stream>>>(
        y1, b2P, b2, NTOK, 128, 512, out, nullptr, x1);
}
